// Decoder_19619410608215
// MI455X (gfx1250) — matrix operand order repaired and checked by reading, not compiled, not run on hardware
//
#include <hip/hip_runtime.h>
#include <hip/hip_bf16.h>
#include <stdint.h>

#define NODES 50000
#define EDGES 800000
#define NREL  8
#define OUTCH 64
#define HID   128

typedef __attribute__((ext_vector_type(16))) __bf16 v16bf;
typedef __attribute__((ext_vector_type(8)))  float  v8f;

__device__ __forceinline__ __bf16 f2bf(float f) { return (__bf16)f; }
__device__ __forceinline__ float  bf2f(__bf16 b) { return (float)b; }
__device__ __forceinline__ unsigned short bfbits(__bf16 b) { return __builtin_bit_cast(unsigned short, b); }
__device__ __forceinline__ __bf16 bitsbf(unsigned short u) { return __builtin_bit_cast(__bf16, u); }

// monotonic sortable key for signed floats (for atomic max of attention logits)
__device__ __forceinline__ unsigned fkey(float f) {
    unsigned u = __float_as_uint(f);
    return (u & 0x80000000u) ? ~u : (u | 0x80000000u);
}
__device__ __forceinline__ float ikey(unsigned k) {
    unsigned u = (k & 0x80000000u) ? (k ^ 0x80000000u) : ~k;
    return __uint_as_float(u);
}
#define KEY_NEG_INF 0x007FFFFFu   // fkey(-inf)

__device__ __forceinline__ void unpack4(uint2 u, float f[4]) {
    f[0] = bf2f(bitsbf((unsigned short)(u.x & 0xffffu)));
    f[1] = bf2f(bitsbf((unsigned short)(u.x >> 16)));
    f[2] = bf2f(bitsbf((unsigned short)(u.y & 0xffffu)));
    f[3] = bf2f(bitsbf((unsigned short)(u.y >> 16)));
}

// ---------------- weight convert + transpose: w[r][k][c] f32 -> wT[r][c][k] bf16 ----
__global__ void cvt_transpose_w_kernel(const float* __restrict__ w,
                                       unsigned short* __restrict__ wT) {
    int idx = blockIdx.x * blockDim.x + threadIdx.x;
    if (idx >= NREL * HID * HID) return;
    int r = idx / (HID * HID), rem = idx % (HID * HID);
    int k = rem / HID, c = rem % HID;
    wT[(size_t)r * HID * HID + c * HID + k] = bfbits(f2bf(w[idx]));
}

// ---------------- h = z @ lin_w.T + lin_b  (M=50000, K=64, N=128), out bf16 --------
// block = 256 threads = 8 waves; wave w owns 16-col tile; block owns 16-row tile.
__global__ void gemm_h_kernel(const float* __restrict__ z,
                              const float* __restrict__ lin_w,   // [128][64]
                              const float* __restrict__ lin_b,
                              unsigned short* __restrict__ h_bf16) {
    const int lane = threadIdx.x & 31;
    const int wave = threadIdx.x >> 5;
    const int hi = lane >> 4, lo = lane & 15;
    const int gm = blockIdx.x * 16 + lo;   // A row held by this lane
    const int gn = wave * 16 + lo;         // B/D column held by this lane

    v8f acc = {};
#pragma unroll
    for (int kc = 0; kc < 2; ++kc) {       // K=64 in two 32-chunks
        v16bf a, b;
#pragma unroll
        for (int v = 0; v < 8; ++v) {
            // A 16x32 bf16 layout: slots 0-3 -> K = hi*8+2v, slots 4-7 -> K = 16+hi*8+2(v-4)
            int ka = kc * 32 + ((v < 4) ? (hi * 8 + 2 * v) : (16 + hi * 8 + 2 * (v - 4)));
            float2 za = *(const float2*)(z + gm * OUTCH + ka);
            a[2 * v]     = f2bf(za.x);
            a[2 * v + 1] = f2bf(za.y);
            // B 32x16 bf16 layout: slots 0-3 -> K = hi*8+2v, slots 4-7 -> K = 16+hi*8+2(v-4)
            int kb = kc * 32 + ((v < 4) ? (hi * 8 + 2 * v) : (16 + hi * 8 + 2 * (v - 4)));
            float2 wb = *(const float2*)(lin_w + gn * OUTCH + kb);  // B[k][n]=lin_w[n][k]
            b[2 * v]     = f2bf(wb.x);
            b[2 * v + 1] = f2bf(wb.y);
        }
        acc = __builtin_amdgcn_wmma_f32_16x16x32_bf16(false, a, false, b,
                                                      (short)0, acc, false, false);
    }
    float bias = lin_b[gn];
#pragma unroll
    for (int v = 0; v < 8; ++v) {
        int row = blockIdx.x * 16 + hi * 8 + v;   // D layout
        h_bf16[row * HID + gn] = bfbits(f2bf(acc[v] + bias));
    }
}

// ---------------- x_all[n,r,:] = h[n,:] @ w[r]  (K=128, N=128), bf16 out -----------
__global__ void gemm_xall_kernel(const unsigned short* __restrict__ h_bf16,
                                 const unsigned short* __restrict__ wT,   // [R][c][k] bf16
                                 unsigned short* __restrict__ x_all) {    // [N][R][128] bf16
    const int r = blockIdx.y;
    const int lane = threadIdx.x & 31;
    const int wave = threadIdx.x >> 5;
    const int hi = lane >> 4, lo = lane & 15;
    const int gm = blockIdx.x * 16 + lo;
    const int gc = wave * 16 + lo;
    const unsigned* hp = (const unsigned*)h_bf16;
    const unsigned* wp = (const unsigned*)(wT + (size_t)r * HID * HID);

    v8f acc = {};
#pragma unroll
    for (int kc = 0; kc < 4; ++kc) {       // K=128 in four 32-chunks
        v16bf a, b;
#pragma unroll
        for (int v = 0; v < 8; ++v) {
            int ka = kc * 32 + ((v < 4) ? (hi * 8 + 2 * v) : (16 + hi * 8 + 2 * (v - 4)));
            unsigned ua = hp[(gm * HID + ka) >> 1];
            a[2 * v]     = bitsbf((unsigned short)(ua & 0xffffu));
            a[2 * v + 1] = bitsbf((unsigned short)(ua >> 16));
            int kb = kc * 32 + ((v < 4) ? (hi * 8 + 2 * v) : (16 + hi * 8 + 2 * (v - 4)));
            unsigned ub = wp[(gc * HID + kb) >> 1];   // wT[c][k] = w[r][k][c]
            b[2 * v]     = bitsbf((unsigned short)(ub & 0xffffu));
            b[2 * v + 1] = bitsbf((unsigned short)(ub >> 16));
        }
        acc = __builtin_amdgcn_wmma_f32_16x16x32_bf16(false, a, false, b,
                                                      (short)0, acc, false, false);
    }
#pragma unroll
    for (int v = 0; v < 8; ++v) {
        int row = blockIdx.x * 16 + hi * 8 + v;
        x_all[((size_t)row * NREL + r) * HID + gc] = bfbits(f2bf(acc[v]));
    }
}

// ---------------- per-edge attention logit + atomic segment max --------------------
// one wave per edge: 256 B row load = 8 B/lane, fully coalesced; shuffle reduction.
__global__ void edge_alpha_kernel(const unsigned short* __restrict__ x_all,
                                  const int* __restrict__ src, const int* __restrict__ dst,
                                  const int* __restrict__ etype,
                                  const float* __restrict__ qv, const float* __restrict__ kv,
                                  float* __restrict__ alpha, unsigned* __restrict__ mkey) {
    const int lane = threadIdx.x & 31;
    const int wid = blockIdx.x * (blockDim.x >> 5) + (threadIdx.x >> 5);
    const int nw = gridDim.x * (blockDim.x >> 5);
    const int c0 = lane * 4;
    float qr[4] = {qv[c0], qv[c0 + 1], qv[c0 + 2], qv[c0 + 3]};
    float kr[4] = {kv[c0], kv[c0 + 1], kv[c0 + 2], kv[c0 + 3]};
    for (int e = wid; e < EDGES; e += nw) {
        int s = src[e], d = dst[e], t = etype[e];
        const unsigned short* xi = x_all + ((size_t)d * NREL + t) * HID;
        const unsigned short* xj = x_all + ((size_t)s * NREL + t) * HID;
        uint2 ui = *(const uint2*)(xi + c0);
        uint2 uj = *(const uint2*)(xj + c0);
        float fi[4], fj[4];
        unpack4(ui, fi);
        unpack4(uj, fj);
        float p = fi[0] * qr[0] + fi[1] * qr[1] + fi[2] * qr[2] + fi[3] * qr[3]
                + fj[0] * kr[0] + fj[1] * kr[1] + fj[2] * kr[2] + fj[3] * kr[3];
#pragma unroll
        for (int off = 16; off > 0; off >>= 1) p += __shfl_xor(p, off, 32);
        if (lane == 0) {
            float a = (p > 0.f) ? p : 0.2f * p;   // leaky relu
            alpha[e] = a;
            atomicMax(&mkey[d], fkey(a));
        }
    }
}

__global__ void node_max_kernel(const unsigned* __restrict__ mkey, float* __restrict__ mval) {
    int n = blockIdx.x * blockDim.x + threadIdx.x;
    if (n >= NODES) return;
    unsigned kk = mkey[n];
    mval[n] = (kk == KEY_NEG_INF) ? 0.f : ikey(kk);   // empty segments -> 0
}

// ---------------- fused exp + denom accumulate + weighted scatter ------------------
__global__ void edge_accum_kernel(const unsigned short* __restrict__ x_all,
                                  const int* __restrict__ src, const int* __restrict__ dst,
                                  const int* __restrict__ etype,
                                  const float* __restrict__ alpha, const float* __restrict__ mval,
                                  float* __restrict__ denom, float* __restrict__ accum) {
    const int lane = threadIdx.x & 31;
    const int wid = blockIdx.x * (blockDim.x >> 5) + (threadIdx.x >> 5);
    const int nw = gridDim.x * (blockDim.x >> 5);
    const int c0 = lane * 4;
    for (int e = wid; e < EDGES; e += nw) {
        int s = src[e], d = dst[e], t = etype[e];
        float ex = __expf(alpha[e] - mval[d]);
        if (lane == 0) atomicAdd(&denom[d], ex);
        const unsigned short* xj = x_all + ((size_t)s * NREL + t) * HID;
        uint2 uj = *(const uint2*)(xj + c0);
        float fj[4];
        unpack4(uj, fj);
        float* ap = accum + (size_t)d * HID + c0;
        atomicAdd(ap + 0, ex * fj[0]);
        atomicAdd(ap + 1, ex * fj[1]);
        atomicAdd(ap + 2, ex * fj[2]);
        atomicAdd(ap + 3, ex * fj[3]);
    }
}

// ---------------- normalize + bias (+relu), emit bf16 (layer1) or f32 (layer2) -----
__global__ void node_finish_kernel(const float* __restrict__ accum,
                                   const float* __restrict__ denom,
                                   const float* __restrict__ bias,
                                   unsigned short* obf, float* of32, int relu) {
    int idx = blockIdx.x * blockDim.x + threadIdx.x;
    if (idx >= NODES * HID) return;
    int n = idx >> 7, c = idx & 127;
    float v = accum[idx] / (denom[n] + 1e-16f) + bias[c];
    if (relu) v = fmaxf(v, 0.f);
    if (obf) obf[idx] = bfbits(f2bf(v));
    if (of32) of32[idx] = v;
}

__global__ void fill_u32_kernel(unsigned* p, unsigned val, int n) {
    int i = blockIdx.x * blockDim.x + threadIdx.x;
    if (i < n) p[i] = val;
}
__global__ void fill_f32_kernel(float* p, float val, int n) {
    int i = blockIdx.x * blockDim.x + threadIdx.x;
    if (i < n) p[i] = val;
}

extern "C" void kernel_launch(void* const* d_in, const int* in_sizes, int n_in,
                              void* d_out, int out_size, void* d_ws, size_t ws_size,
                              hipStream_t stream) {
    (void)in_sizes; (void)n_in; (void)out_size; (void)ws_size;
    const float* z     = (const float*)d_in[0];
    const float* lin_w = (const float*)d_in[1];
    const float* lin_b = (const float*)d_in[2];
    const float* w1    = (const float*)d_in[3];
    const float* q1    = (const float*)d_in[4];
    const float* k1    = (const float*)d_in[5];
    const float* b1    = (const float*)d_in[6];
    const float* w2    = (const float*)d_in[7];
    const float* q2    = (const float*)d_in[8];
    const float* k2    = (const float*)d_in[9];
    const float* b2    = (const float*)d_in[10];
    const int* ei      = (const int*)d_in[11];
    const int* etype   = (const int*)d_in[12];
    const int* esrc = ei;
    const int* edst = ei + EDGES;
    float* outp = (float*)d_out;

    // workspace carve-out (256B-aligned), ~145 MB total
    char* ws = (char*)d_ws;
    size_t off = 0;
    auto take = [&](size_t bytes) -> char* {
        char* p = ws + off;
        off = (off + bytes + 255) & ~(size_t)255;
        return p;
    };
    unsigned short* w1T   = (unsigned short*)take((size_t)NREL * HID * HID * 2);
    unsigned short* w2T   = (unsigned short*)take((size_t)NREL * HID * HID * 2);
    unsigned short* hbf   = (unsigned short*)take((size_t)NODES * HID * 2);
    unsigned short* xall  = (unsigned short*)take((size_t)NODES * NREL * HID * 2); // 102 MB, fits L2
    float*    alpha  = (float*)take((size_t)EDGES * 4);
    unsigned* mkey   = (unsigned*)take((size_t)NODES * 4);
    float*    mval   = (float*)take((size_t)NODES * 4);
    float*    denom  = (float*)take((size_t)NODES * 4);
    float*    accum1 = (float*)take((size_t)NODES * HID * 4);

    const int wn = NREL * HID * HID;
    cvt_transpose_w_kernel<<<(wn + 255) / 256, 256, 0, stream>>>(w1, w1T);
    cvt_transpose_w_kernel<<<(wn + 255) / 256, 256, 0, stream>>>(w2, w2T);

    gemm_h_kernel<<<NODES / 16, 256, 0, stream>>>(z, lin_w, lin_b, hbf);

    auto run_layer = [&](const unsigned short* wT, const float* qv, const float* kv,
                         const float* bias, float* accum, unsigned short* obf,
                         float* of32, int relu) {
        gemm_xall_kernel<<<dim3(NODES / 16, NREL), 256, 0, stream>>>(hbf, wT, xall);
        fill_u32_kernel<<<(NODES + 255) / 256, 256, 0, stream>>>(mkey, KEY_NEG_INF, NODES);
        fill_f32_kernel<<<(NODES + 255) / 256, 256, 0, stream>>>(denom, 0.f, NODES);
        fill_f32_kernel<<<(NODES * HID + 255) / 256, 256, 0, stream>>>(accum, 0.f, NODES * HID);
        edge_alpha_kernel<<<4096, 256, 0, stream>>>(xall, esrc, edst, etype, qv, kv, alpha, mkey);
        node_max_kernel<<<(NODES + 255) / 256, 256, 0, stream>>>(mkey, mval);
        edge_accum_kernel<<<4096, 256, 0, stream>>>(xall, esrc, edst, etype, alpha, mval, denom, accum);
        node_finish_kernel<<<(NODES * HID + 255) / 256, 256, 0, stream>>>(accum, denom, bias, obf, of32, relu);
    };

    // layer 1: relu, result back into hbf (bf16) as input to layer 2
    run_layer(w1T, q1, k1, b1, accum1, hbf, nullptr, 1);
    // layer 2: accumulate directly into d_out, finish in place (f32)
    run_layer(w2T, q2, k2, b2, outp, nullptr, outp, 0);
}
